// GAT_43997644981193
// MI455X (gfx1250) — matrix operand order repaired and checked by reading, not compiled, not run on hardware
//
#include <hip/hip_runtime.h>
#include <math.h>

// GAT layer for MI455X (gfx1250, wave32).
// Sizes from the reference: N=50000, E=1.6e6, DIN=256, DOUT=128, H=4, S=32.
#define DIN  256
#define DOUT 128
#define NH   4
#define SH   32   // DOUT / NH

typedef __attribute__((ext_vector_type(16))) _Float16 v16h;
typedef __attribute__((ext_vector_type(8)))  float    v8f;

// ---------------------------------------------------------------------------
// Kernel 1: build CSR row offsets from the sorted `row` array (lower_bound).
// ---------------------------------------------------------------------------
__global__ void gat_build_row_ptr(const int* __restrict__ row,
                                  int* __restrict__ row_ptr, int N, int E) {
  int n = blockIdx.x * blockDim.x + threadIdx.x;
  if (n > N) return;
  if (n == N) { row_ptr[N] = E; return; }
  int lo = 0, hi = E;
  while (lo < hi) {                 // first e with row[e] >= n
    int mid = (lo + hi) >> 1;
    if (row[mid] < n) lo = mid + 1; else hi = mid;
  }
  row_ptr[n] = lo;
}

// ---------------------------------------------------------------------------
// Kernel 2: fs = relu(feat @ W0^T + b0), fn = relu(feat @ W1^T + b1)
// One wave32 per 16-node tile. f16 WMMA, f32 accumulate.
//   A (16x32 f16)  lane<16 : M=lane,    halves 0..7 -> K 0..7,  8..15 -> K 16..23
//                  lane>=16: M=lane-16, halves 0..7 -> K 8..15, 8..15 -> K 24..31
//   B (32x16 f16)  lane<16 : N=lane,    halves 0..7 -> K 0..7,  8..15 -> K 16..23
//                  lane>=16: N=lane-16, halves 0..7 -> K 8..15, 8..15 -> K 24..31
//   C (16x16 f32)  lane<16 : N=lane, vgpr r -> M=r ;  lane>=16: M=r+8
// ---------------------------------------------------------------------------
__global__ __launch_bounds__(256) void gat_gemm_act(
    const float* __restrict__ feat,
    const float* __restrict__ W0, const float* __restrict__ b0,
    const float* __restrict__ W1, const float* __restrict__ b1,
    float* __restrict__ fs, float* __restrict__ fn, int nTiles) {
  const int lane = threadIdx.x & 31;
  const int tile = (blockIdx.x * blockDim.x + threadIdx.x) >> 5;
  if (tile >= nTiles) return;        // wave-uniform: EXEC all-ones inside
  const int mrow  = lane & 15;
  const int kg    = lane >> 4;
  const int node0 = tile * 16;
  const float* frow = feat + (size_t)(node0 + mrow) * DIN;

  // Load the full 16x256 A tile once (8 fragments, 64 VGPRs as f16).
  v16h A[8];
#pragma unroll
  for (int kc = 0; kc < 8; ++kc) {
    const int ka = kc * 32 + (kg ? 8 : 0);     // halves 0..7  (8 consecutive)
    const int kb = kc * 32 + (kg ? 24 : 16);   // halves 8..15 (8 consecutive)
#pragma unroll
    for (int i = 0; i < 8; ++i) A[kc][i]     = (_Float16)frow[ka + i];
#pragma unroll
    for (int i = 0; i < 8; ++i) A[kc][8 + i] = (_Float16)frow[kb + i];
  }

  const float* Ws[2]  = {W0, W1};
  const float* bias[2]= {b0, b1};
  float*       Os[2]  = {fs, fn};

#pragma unroll
  for (int m = 0; m < 2; ++m) {
    const float* W = Ws[m];
    for (int c = 0; c < 8; ++c) {              // 8 output-column tiles of 16
      const int ncol = c * 16 + mrow;          // output channel == W row
      const float* wrow = W + (size_t)ncol * DIN + (kg ? 8 : 0);
      v8f acc = {};
#pragma unroll
      for (int kc = 0; kc < 8; ++kc) {
        v16h B;
#pragma unroll
        for (int i = 0; i < 16; ++i)           // two runs of 8 f32 -> f16 (table order)
          B[i] = (_Float16)wrow[kc * 32 + (i < 8 ? i : i + 8)];
        acc = __builtin_amdgcn_wmma_f32_16x16x32_f16(
            false, A[kc], false, B, (short)0, acc, false, false);
      }
      const float bv = bias[m][ncol];
      float* out = Os[m];
#pragma unroll
      for (int r = 0; r < 8; ++r) {
        const int mr = r + (kg ? 8 : 0);
        float v = acc[r] + bv;
        out[(size_t)(node0 + mr) * DOUT + ncol] = v > 0.f ? v : 0.f;
      }
    }
  }
}

// ---------------------------------------------------------------------------
// Kernel 3: a_s[n,h] = leaky(dot(att[0,h,:], fs[n,h,:]), 0.2), same for a_n.
// ---------------------------------------------------------------------------
__global__ void gat_attn(const float* __restrict__ fs,
                         const float* __restrict__ fn,
                         const float* __restrict__ att,
                         float* __restrict__ a_s, float* __restrict__ a_n,
                         int N) {
  int t = blockIdx.x * blockDim.x + threadIdx.x;
  if (t >= N * NH) return;
  const int node = t >> 2, h = t & 3;
  const float* a0 = att + h * SH;          // attention[0][h][:]
  const float* a1 = att + DOUT + h * SH;   // attention[1][h][:]
  const float* x0 = fs + (size_t)node * DOUT + h * SH;
  const float* x1 = fn + (size_t)node * DOUT + h * SH;
  float s0 = 0.f, s1 = 0.f;
#pragma unroll
  for (int s = 0; s < SH; ++s) { s0 += a0[s] * x0[s]; s1 += a1[s] * x1[s]; }
  a_s[t] = s0 > 0.f ? s0 : 0.2f * s0;
  a_n[t] = s1 > 0.f ? s1 : 0.2f * s1;
}

// ---------------------------------------------------------------------------
// Kernel 4: one wave32 per destination node.
//   lane -> head h = lane>>3, channels sb = (lane&7)*4 .. +3  (4 floats/lane)
//   pass 1: per-head segment max (edge-parallel + full-wave shfl_xor reduce)
//   pass 2: softmax weights + weighted gather of fn[col] (wave walks edges,
//           each lane owns 4 channels) -> agg
//   epilogue: per-head layernorm of fs and agg via width-8 shfl groups.
// No atomics anywhere.
// ---------------------------------------------------------------------------
__global__ __launch_bounds__(256) void gat_node_fuse(
    const int* __restrict__ col, const int* __restrict__ row_ptr,
    const float* __restrict__ edge_val,
    const float* __restrict__ fs, const float* __restrict__ fn,
    const float* __restrict__ a_s, const float* __restrict__ a_n,
    const float* __restrict__ scale, const float* __restrict__ offset,
    float* __restrict__ out, int N) {
  const int lane = threadIdx.x & 31;
  const int node = (blockIdx.x * blockDim.x + threadIdx.x) >> 5;
  if (node >= N) return;             // wave-uniform
  const int h  = lane >> 3;
  const int sb = (lane & 7) * 4;
  const int e0 = row_ptr[node], e1 = row_ptr[node + 1];
  const float4 as4 = *(const float4*)(a_s + (size_t)node * NH);

  // ---- pass 1: segment max over all 4 heads, edges strided across lanes ----
  float4 m4 = make_float4(-INFINITY, -INFINITY, -INFINITY, -INFINITY);
  for (int e = e0 + lane; e < e1; e += 32) {
    const int c = col[e];
    const float4 an4 = *(const float4*)(a_n + (size_t)c * NH);
    m4.x = fmaxf(m4.x, as4.x + an4.x);
    m4.y = fmaxf(m4.y, as4.y + an4.y);
    m4.z = fmaxf(m4.z, as4.z + an4.z);
    m4.w = fmaxf(m4.w, as4.w + an4.w);
    // warm L2/L0 for pass 2's gather of this fn row (global_prefetch_b8)
    __builtin_prefetch(fn + (size_t)c * DOUT + h * SH + sb, 0, 0);
  }
#pragma unroll
  for (int d = 16; d; d >>= 1) {
    m4.x = fmaxf(m4.x, __shfl_xor(m4.x, d, 32));
    m4.y = fmaxf(m4.y, __shfl_xor(m4.y, d, 32));
    m4.z = fmaxf(m4.z, __shfl_xor(m4.z, d, 32));
    m4.w = fmaxf(m4.w, __shfl_xor(m4.w, d, 32));
  }
  const float mh  = (h == 0) ? m4.x : (h == 1) ? m4.y : (h == 2) ? m4.z : m4.w;
  const float ash = (h == 0) ? as4.x : (h == 1) ? as4.y : (h == 2) ? as4.z : as4.w;

  // ---- pass 2: softmax-weighted aggregation; lane owns 4 output channels ---
  float ax = 0.f, ay = 0.f, az = 0.f, aw = 0.f, dsum = 0.f;
  for (int e = e0; e < e1; ++e) {
    const int c = col[e];
    const float w = __expf(ash + a_n[(size_t)c * NH + h] - mh) * edge_val[e];
    dsum += w;
    const float4 f = *(const float4*)(fn + (size_t)c * DOUT + h * SH + sb);
    ax += w * f.x; ay += w * f.y; az += w * f.z; aw += w * f.w;
  }
  const float inv = 1.f / fmaxf(dsum, 1e-10f);   // clip(denom, 1e-10)
  float4 agg = make_float4(ax * inv, ay * inv, az * inv, aw * inv);
  const float4 fs4 = *(const float4*)(fs + (size_t)node * DOUT + h * SH + sb);

  // ---- per-head layernorm over 32 channels = one 8-lane group --------------
  auto lnorm = [&](float4 v, const float* sc, const float* of) -> float4 {
    float s = v.x + v.y + v.z + v.w;
    s += __shfl_xor(s, 1, 8); s += __shfl_xor(s, 2, 8); s += __shfl_xor(s, 4, 8);
    const float mean = s * (1.f / 32.f);
    const float dx = v.x - mean, dy = v.y - mean, dz = v.z - mean, dw = v.w - mean;
    float q = dx * dx + dy * dy + dz * dz + dw * dw;
    q += __shfl_xor(q, 1, 8); q += __shfl_xor(q, 2, 8); q += __shfl_xor(q, 4, 8);
    const float r = rsqrtf(q * (1.f / 32.f) + 1e-9f);
    return make_float4(dx * r * sc[sb + 0] + of[sb + 0],
                       dy * r * sc[sb + 1] + of[sb + 1],
                       dz * r * sc[sb + 2] + of[sb + 2],
                       dw * r * sc[sb + 3] + of[sb + 3]);
  };
  const float4 sbn = lnorm(fs4, scale + h * SH,        offset + h * SH);
  const float4 nbn = lnorm(agg, scale + DOUT + h * SH, offset + DOUT + h * SH);

  float4 o = make_float4((sbn.x + nbn.x) * 0.5f, (sbn.y + nbn.y) * 0.5f,
                         (sbn.z + nbn.z) * 0.5f, (sbn.w + nbn.w) * 0.5f);
  *(float4*)(out + (size_t)node * DOUT + h * SH + sb) = o;
}

// ---------------------------------------------------------------------------
extern "C" void kernel_launch(void* const* d_in, const int* in_sizes, int n_in,
                              void* d_out, int out_size, void* d_ws, size_t ws_size,
                              hipStream_t stream) {
  const float* feat     = (const float*)d_in[0];
  const int*   row      = (const int*)  d_in[1];
  const int*   col      = (const int*)  d_in[2];
  const float* edge_val = (const float*)d_in[3];
  const float* W0       = (const float*)d_in[4];
  const float* b0       = (const float*)d_in[5];
  const float* W1       = (const float*)d_in[6];
  const float* b1       = (const float*)d_in[7];
  const float* att      = (const float*)d_in[8];   // [2,4,32]
  const float* scale    = (const float*)d_in[9];   // [2,4,32]
  const float* offset   = (const float*)d_in[10];  // [2,4,32]
  const int N = in_sizes[0] / DIN;
  const int E = in_sizes[1];

  // workspace layout (~53 MB): fs | fn | a_s | a_n | row_ptr
  float* fs    = (float*)d_ws;
  float* fn    = fs  + (size_t)N * DOUT;
  float* a_s   = fn  + (size_t)N * DOUT;
  float* a_n   = a_s + (size_t)N * NH;
  int*   rptr  = (int*)(a_n + (size_t)N * NH);
  float* out   = (float*)d_out;

  // 1) CSR offsets from sorted row (binary search per node)
  gat_build_row_ptr<<<(N + 1 + 255) / 256, 256, 0, stream>>>(row, rptr, N, E);

  // 2) WMMA GEMMs + bias + relu (one wave per 16-node tile, 8 waves/block)
  const int nTiles = (N + 15) / 16;
  gat_gemm_act<<<(nTiles + 7) / 8, 256, 0, stream>>>(feat, W0, b0, W1, b1,
                                                     fs, fn, nTiles);

  // 3) attention scores + leaky relu
  gat_attn<<<(N * NH + 255) / 256, 256, 0, stream>>>(fs, fn, att, a_s, a_n, N);

  // 4) fused per-node softmax / aggregate / layernorm / output (1 wave/node)
  gat_node_fuse<<<(N + 7) / 8, 256, 0, stream>>>(col, rptr, edge_val, fs, fn,
                                                 a_s, a_n, scale, offset, out, N);
}
